// E71MatrixGatedTritonCell_55456617726101
// MI455X (gfx1250) — compile-verified, matrix operand order checked by reading, not run on hardware
//
#include <hip/hip_runtime.h>

typedef __attribute__((ext_vector_type(16))) __bf16 v16bf;
typedef __attribute__((ext_vector_type(8)))  float  v8f;
typedef __attribute__((ext_vector_type(4)))  float  f32x4;

#define T_STEPS 512
#define BATCH   32
#define DIM     1024
#define NS      64
#define M_TOTAL (T_STEPS * BATCH)          // 16384 rows
#define KTILES  (DIM / 32)                 // 32 k-steps
#define PROJ_ELEMS ((size_t)4 * M_TOTAL * NS)

// ---------------------------------------------------------------------------
// Phase 0: pre-swizzle the 4 weight matrices into bf16, stored in the exact
// WMMA B-fragment order: fragment (p, nt, kt) = 32 lanes x 16 bf16 (32 B per
// lane, contiguous).  In the GEMM each lane then loads its whole B fragment
// with one aligned 32-byte vector load: no converts, no v_mov_b16 shuffles.
//   lane L (<16):  N = nt*16 + L,      K = kt*32 + {0..7, 16..23}
//   lane L (>=16): N = nt*16 + (L-16), K = kt*32 + {8..15, 24..31}
// ---------------------------------------------------------------------------
__global__ __launch_bounds__(256)
void weight_swizzle_kernel(const float* __restrict__ Wk,
                           const float* __restrict__ Wv,
                           const float* __restrict__ Wq,
                           const float* __restrict__ Wa,
                           __bf16* __restrict__ wbf)
{
    const int tid  = blockIdx.x * 256 + threadIdx.x;   // 16384 total
    const int lane = tid & 31;
    const int kt   = (tid >> 5) & (KTILES - 1);
    const int nt   = (tid >> 10) & 3;
    const int p    = (tid >> 12) & 3;

    const float* const Wmat[4] = {Wk, Wv, Wq, Wa};
    const float* __restrict__ W = Wmat[p];

    const int n     = nt * 16 + (lane & 15);
    const int khalf = (lane >> 4) << 3;                 // 0 or 8
    const float* __restrict__ wrow = W + (size_t)n * DIM + kt * 32 + khalf;

    v16bf frag;
    #pragma unroll
    for (int j = 0; j < 8; ++j) {
        frag[j]     = (__bf16)wrow[j];
        frag[8 + j] = (__bf16)wrow[16 + j];
    }

    v16bf* __restrict__ dst =
        (v16bf*)(wbf + ((size_t)(((p * 4 + nt) * KTILES + kt) * 32 + lane)) * 16);
    *dst = frag;
}

// ---------------------------------------------------------------------------
// Phase 1: fused 4-way projection GEMM.
//   proj[p][m][n] = sum_d x[m][d] * W_p[n][d]      (p = k,v,q,alpha)
// One block per 128-row M tile; each wave owns 16 M-rows and computes ALL
// FOUR projections (16 WMMAs per k-step) so x is streamed from HBM exactly
// once.  A converted fp32->bf16 in-register; B loaded pre-swizzled bf16.
// ---------------------------------------------------------------------------
__global__ __launch_bounds__(256)
void proj_gemm_kernel(const float* __restrict__ x,
                      const __bf16* __restrict__ wbf,
                      float* __restrict__ proj)
{
    const int wave  = threadIdx.x >> 5;
    const int lane  = threadIdx.x & 31;
    const int lmod  = lane & 15;
    const int khalf = (lane >> 4) << 3;          // 0 or 8

    const int mbase = blockIdx.x * 128 + wave * 16;
    const float* __restrict__ xrow = x + (size_t)(mbase + lmod) * DIM;

    // per-lane base into the swizzled weights; fragment (p,nt,kt) stride = 512
    const v16bf* __restrict__ bbase = (const v16bf*)(wbf) + lane;

    v8f c[16];                                   // [p][nt], 128 VGPRs
    #pragma unroll
    for (int i = 0; i < 16; ++i) c[i] = (v8f){};

    for (int kt = 0; kt < KTILES; ++kt) {
        const int k0 = kt * 32;
        // prefetch next A chunk (global_prefetch_b8)
        if (kt + 2 < KTILES) __builtin_prefetch(xrow + k0 + 64, 0, 0);

        // ---- A fragment: 16x32 bf16 per ISA layout (shared by all 16 WMMAs)
        const f32x4* __restrict__ xa = (const f32x4*)(xrow + k0 + khalf);
        const f32x4* __restrict__ xb = (const f32x4*)(xrow + k0 + 16 + khalf);
        f32x4 a0 = xa[0], a1 = xa[1], a2 = xb[0], a3 = xb[1];
        v16bf afrag;
        #pragma unroll
        for (int j = 0; j < 4; ++j) {
            afrag[j]      = (__bf16)a0[j];
            afrag[4 + j]  = (__bf16)a1[j];
            afrag[8 + j]  = (__bf16)a2[j];
            afrag[12 + j] = (__bf16)a3[j];
        }

        #pragma unroll
        for (int p = 0; p < 4; ++p) {
            // Load all 4 pre-swizzled B fragments (one 32-B vector load each),
            // then issue the 4 WMMAs back-to-back.
            v16bf bfrag[4];
            #pragma unroll
            for (int nt = 0; nt < 4; ++nt) {
                bfrag[nt] = bbase[(size_t)((p * 4 + nt) * KTILES + kt) * 32];
            }
            #pragma unroll
            for (int nt = 0; nt < 4; ++nt) {
                c[p * 4 + nt] = __builtin_amdgcn_wmma_f32_16x16x32_bf16(
                    false, afrag, false, bfrag[nt], (short)0, c[p * 4 + nt],
                    false, false);
            }
        }
    }

    // ---- store: C/D layout -> element r is (M = r + 8*(lane/16), N = lane%16)
    const int mstore = mbase + ((lane >> 4) << 3);
    #pragma unroll
    for (int p = 0; p < 4; ++p) {
        float* __restrict__ outp = proj + (size_t)p * M_TOTAL * NS;
        #pragma unroll
        for (int nt = 0; nt < 4; ++nt) {
            #pragma unroll
            for (int r = 0; r < 8; ++r) {
                outp[(size_t)(mstore + r) * NS + nt * 16 + lmod] = c[p * 4 + nt][r];
            }
        }
    }
}

// ---------------------------------------------------------------------------
// Phase 2: sequential gated scan.  One block per batch element (32 blocks),
// 64 threads = 2 wave32s.  Thread i owns row i of S in 64 VGPRs.
// ---------------------------------------------------------------------------
__global__ __launch_bounds__(64)
void scan_kernel(const float* __restrict__ proj,
                 const float* __restrict__ S0,
                 const float* __restrict__ d_alpha,
                 const float* __restrict__ b_alpha,
                 float* __restrict__ out,      // [T, B, NS]
                 float* __restrict__ Sfinal)   // [B, NS, NS]
{
    const int b = blockIdx.x;
    const int i = threadIdx.x;                  // state row 0..63

    const float* __restrict__ kp = proj;
    const float* __restrict__ vp = proj + (size_t)1 * M_TOTAL * NS;
    const float* __restrict__ qp = proj + (size_t)2 * M_TOTAL * NS;
    const float* __restrict__ ap = proj + (size_t)3 * M_TOTAL * NS;

    float Srow[NS];
    #pragma unroll
    for (int j = 0; j < NS; ++j)
        Srow[j] = S0[(size_t)b * NS * NS + (size_t)i * NS + j];

    const float da = d_alpha[i];
    const float ba = b_alpha[i];

    __shared__ float ks[NS];
    __shared__ float qs[NS];

    for (int t = 0; t < T_STEPS; ++t) {
        const size_t off = ((size_t)t * BATCH + b) * NS;
        const float kv = kp[off + i];
        const float qv = qp[off + i];
        const float vi = vp[off + i];
        const float ai = ap[off + i];

        __syncthreads();               // previous-iteration reads done
        ks[i] = kv;
        qs[i] = qv;
        __syncthreads();

        // retrieved = S @ k   (4 accumulators for ILP)
        float r0 = 0.f, r1 = 0.f, r2 = 0.f, r3 = 0.f;
        #pragma unroll
        for (int j = 0; j < NS; j += 4) {
            r0 = __builtin_fmaf(Srow[j + 0], ks[j + 0], r0);
            r1 = __builtin_fmaf(Srow[j + 1], ks[j + 1], r1);
            r2 = __builtin_fmaf(Srow[j + 2], ks[j + 2], r2);
            r3 = __builtin_fmaf(Srow[j + 3], ks[j + 3], r3);
        }
        const float retrieved = (r0 + r1) + (r2 + r3);

        const float z     = ai + da * retrieved + ba;
        const float alpha = 1.0f / (1.0f + __expf(-z));
        const float beta  = (1.0f - alpha) * vi;   // (1-alpha) * v_i

        // S_row = alpha*S_row + beta*k ;  h = S_row . q   (fused)
        float h0 = 0.f, h1 = 0.f, h2 = 0.f, h3 = 0.f;
        #pragma unroll
        for (int j = 0; j < NS; j += 4) {
            float s0 = __builtin_fmaf(alpha, Srow[j + 0], beta * ks[j + 0]);
            float s1 = __builtin_fmaf(alpha, Srow[j + 1], beta * ks[j + 1]);
            float s2 = __builtin_fmaf(alpha, Srow[j + 2], beta * ks[j + 2]);
            float s3 = __builtin_fmaf(alpha, Srow[j + 3], beta * ks[j + 3]);
            Srow[j + 0] = s0; Srow[j + 1] = s1;
            Srow[j + 2] = s2; Srow[j + 3] = s3;
            h0 = __builtin_fmaf(s0, qs[j + 0], h0);
            h1 = __builtin_fmaf(s1, qs[j + 1], h1);
            h2 = __builtin_fmaf(s2, qs[j + 2], h2);
            h3 = __builtin_fmaf(s3, qs[j + 3], h3);
        }
        const float h = (h0 + h1) + (h2 + h3);
        // out = h * silu(h) = h^2 * sigmoid(h)
        out[off + i] = h * h * (1.0f / (1.0f + __expf(-h)));
    }

    #pragma unroll
    for (int j = 0; j < NS; ++j)
        Sfinal[(size_t)b * NS * NS + (size_t)i * NS + j] = Srow[j];
}

// ---------------------------------------------------------------------------
extern "C" void kernel_launch(void* const* d_in, const int* in_sizes, int n_in,
                              void* d_out, int out_size, void* d_ws, size_t ws_size,
                              hipStream_t stream)
{
    (void)in_sizes; (void)n_in; (void)out_size; (void)ws_size;

    const float* x   = (const float*)d_in[0];   // [T,B,DIM]
    const float* S0  = (const float*)d_in[1];   // [B,NS,NS]
    const float* Wk  = (const float*)d_in[2];   // [NS,DIM]
    const float* Wv  = (const float*)d_in[3];
    const float* Wq  = (const float*)d_in[4];
    const float* Wa  = (const float*)d_in[5];
    const float* dal = (const float*)d_in[6];   // [NS]
    const float* bal = (const float*)d_in[7];   // [NS]

    float* out    = (float*)d_out;                               // [T,B,NS]
    float* Sfinal = out + (size_t)T_STEPS * BATCH * NS;          // [B,NS,NS]

    float*  proj = (float*)d_ws;                                 // 16 MB
    __bf16* wbf  = (__bf16*)((char*)d_ws + PROJ_ELEMS * sizeof(float)); // 512 KB

    // Phase 0: swizzle weights to bf16 fragment layout (16384 threads)
    weight_swizzle_kernel<<<dim3(64), 256, 0, stream>>>(Wk, Wv, Wq, Wa, wbf);

    // Phase 1: projections
    proj_gemm_kernel<<<dim3(M_TOTAL / 128), 256, 0, stream>>>(x, wbf, proj);

    // Phase 2: recurrence
    scan_kernel<<<BATCH, 64, 0, stream>>>(proj, S0, dal, bal, out, Sfinal);
}
